// Cross_Attention_Layer_73126113182198
// MI455X (gfx1250) — matrix operand order repaired and checked by reading, not compiled, not run on hardware
//
#include <hip/hip_runtime.h>

typedef unsigned short u16;
typedef unsigned int   u32;

typedef __attribute__((ext_vector_type(16))) __bf16 v16bf;
typedef __attribute__((ext_vector_type(8)))  float  v8f;

#define DEV __device__ __forceinline__

static constexpr int       Bb  = 8;
static constexpr int       Nn  = 2048;
static constexpr int       Dd  = 1024;
static constexpr long long BND = (long long)Bb * Nn * Dd;   // 16,777,216
static constexpr long long BNN = (long long)Bb * Nn * Nn;   // 33,554,432

// ---------- fp32 <-> bf16 split helpers (round-to-nearest-even) ----------
DEV u16   f2bf(float f) { u32 u = __float_as_uint(f); u += 0x7FFFu + ((u >> 16) & 1u); return (u16)(u >> 16); }
DEV float bf2f(u16 h)   { return __uint_as_float(((u32)h) << 16); }
// order-preserving float->uint key for atomicMax on floats (incl. negatives)
DEV u32   fkey(float f) { u32 u = __float_as_uint(f); return (u & 0x80000000u) ? ~u : (u | 0x80000000u); }
DEV float fdec(u32 k)   { u32 u = (k & 0x80000000u) ? (k & 0x7FFFFFFFu) : ~k; return __uint_as_float(u); }

// ---------- elementwise kernels ----------
__global__ void split_kernel(const float* __restrict__ x, u16* __restrict__ hi,
                             u16* __restrict__ lo, long long n) {
    long long i = (long long)blockIdx.x * blockDim.x + threadIdx.x;
    long long stride = (long long)gridDim.x * blockDim.x;
    for (; i < n; i += stride) {
        float v = x[i];
        u16 h = f2bf(v);
        hi[i] = h;
        lo[i] = f2bf(v - bf2f(h));
    }
}

__global__ void fill_u32(u32* __restrict__ p, u32 v, int n) {
    int i = blockIdx.x * blockDim.x + threadIdx.x;
    if (i < n) p[i] = v;
}

// ---------- WMMA fragment loaders (CDNA5 bf16 16x16x32 layouts) ----------
union FragBF { v16bf v; uint4 q[2]; };
union AccF   { v8f v; float f[8]; };

// A (16x32, row-major source): lane h=lane/16,r=lane%16 holds row r,
// elems 0..7 = K[k0+h*8 ..], elems 8..15 = K[k0+16+h*8 ..]  (two 16B chunks)
DEV v16bf frag_a(const u16* base, int ld, int row0, int k0, int lane) {
    int h = lane >> 4, r = lane & 15;
    const u16* p = base + (size_t)(row0 + r) * ld + k0 + h * 8;
    FragBF f;
    f.q[0] = *(const uint4*)p;
    f.q[1] = *(const uint4*)(p + 16);
    return f.v;
}
// B (32x16): lane holds column c=lane%16; elems 0..7 = K[k0+h*8 ..],
// elems 8..15 = K[k0+16+h*8 ..] of source row c  (two 16B chunks)
DEV v16bf frag_b(const u16* base, int ld, int col0, int k0, int lane) {
    int h = lane >> 4, c = lane & 15;
    const u16* p = base + (size_t)(col0 + c) * ld + k0 + h * 8;
    FragBF f;
    f.q[0] = *(const uint4*)p;
    f.q[1] = *(const uint4*)(p + 16);
    return f.v;
}

DEV v8f wmma_bf16(v16bf a, v16bf b, v8f c) {
    return __builtin_amdgcn_wmma_f32_16x16x32_bf16(false, a, false, b, (short)0, c, false, false);
}
// bf16x3: full product ~ hi*hi + hi*lo + lo*hi (fp32 accumulate)
DEV v8f mm3(v16bf ah, v16bf al, v16bf bh, v16bf bl, v8f c) {
    c = wmma_bf16(ah, bh, c);
    c = wmma_bf16(ah, bl, c);
    c = wmma_bf16(al, bh, c);
    return c;
}

// ---------- unified bf16x3 GEMM (32x32 per wave, 128x64 per block) ----------
enum { MODE_SPLIT = 0, MODE_SPLIT_T = 1, MODE_SCORE = 2, MODE_FINAL = 3 };

template <int MODE>
__global__ void __launch_bounds__(256)
gemm_bf16x3(const u16* __restrict__ Ah, const u16* __restrict__ Al, int lda, long long sA,
            const u16* __restrict__ Bh, const u16* __restrict__ Bl, int ldb, long long sB,
            int Kdim, int ldo, long long sO,
            float* __restrict__ fout, u16* __restrict__ oh, u16* __restrict__ ol,
            u32* __restrict__ cmax, const float* __restrict__ addin, float* __restrict__ feat)
{
    const int lane = threadIdx.x & 31;
    const int wv   = threadIdx.x >> 5;
    const int row0 = blockIdx.x * 128 + (wv & 3) * 32;
    const int col0 = blockIdx.y * 64  + (wv >> 2) * 32;
    const int bz   = blockIdx.z;
    Ah += (size_t)bz * sA; Al += (size_t)bz * sA;
    Bh += (size_t)bz * sB; Bl += (size_t)bz * sB;

    v8f acc[2][2] = {};
    for (int k0 = 0; k0 < Kdim; k0 += 32) {
        v16bf a0h = frag_a(Ah, lda, row0,      k0, lane);
        v16bf a0l = frag_a(Al, lda, row0,      k0, lane);
        v16bf a1h = frag_a(Ah, lda, row0 + 16, k0, lane);
        v16bf a1l = frag_a(Al, lda, row0 + 16, k0, lane);
        v16bf b0h = frag_b(Bh, ldb, col0,      k0, lane);
        v16bf b0l = frag_b(Bl, ldb, col0,      k0, lane);
        v16bf b1h = frag_b(Bh, ldb, col0 + 16, k0, lane);
        v16bf b1l = frag_b(Bl, ldb, col0 + 16, k0, lane);
        acc[0][0] = mm3(a0h, a0l, b0h, b0l, acc[0][0]);
        acc[0][1] = mm3(a0h, a0l, b1h, b1l, acc[0][1]);
        acc[1][0] = mm3(a1h, a1l, b0h, b0l, acc[1][0]);
        acc[1][1] = mm3(a1h, a1l, b1h, b1l, acc[1][1]);
    }

    float cm[2] = { -3.4e38f, -3.4e38f };
    #pragma unroll
    for (int i = 0; i < 2; i++)
        #pragma unroll
        for (int j = 0; j < 2; j++) {
            AccF u; u.v = acc[i][j];
            #pragma unroll
            for (int e = 0; e < 8; e++) {
                float v = u.f[e];
                int r = row0 + i * 16 + (lane >> 4) * 8 + e;  // C/D: VGPR e -> M=(lane/16)*8+e
                int c = col0 + j * 16 + (lane & 15);          //      N = lane%16
                if (MODE == MODE_SPLIT) {
                    size_t idx = (size_t)r * ldo + c;
                    u16 h = f2bf(v);
                    oh[idx] = h; ol[idx] = f2bf(v - bf2f(h));
                } else if (MODE == MODE_SPLIT_T) {            // store V transposed: Vt[b][v][m]
                    int batch = r >> 11; int m = r & 2047;
                    size_t idx = ((size_t)batch * Dd + c) * Nn + m;
                    u16 h = f2bf(v);
                    oh[idx] = h; ol[idx] = f2bf(v - bf2f(h));
                } else if (MODE == MODE_SCORE) {
                    size_t idx = (size_t)bz * sO + (size_t)r * ldo + c;
                    fout[idx] = v;
                    cm[j] = fmaxf(cm[j], v);
                } else {                                      // MODE_FINAL: out + feature
                    size_t idx = ((size_t)bz * Nn + r) * Dd + c;
                    fout[idx] = v;
                    feat[idx] = v + addin[idx];
                }
            }
        }

    if (MODE == MODE_SCORE) {
        #pragma unroll
        for (int j = 0; j < 2; j++) {
            float m = fmaxf(cm[j], __shfl_xor(cm[j], 16));    // lanes l and l+16 share columns
            if (lane < 16)
                atomicMax(&cmax[(size_t)bz * Nn + col0 + j * 16 + lane], fkey(m));
        }
    }
}

// ---------- per-column (over n) sum of exp, write 1/sum and decoded max ----------
__global__ void __launch_bounds__(256)
colsum_kernel(const float* __restrict__ S, u32* __restrict__ keys, float* __restrict__ colrs) {
    int b  = blockIdx.x >> 5;
    int m0 = (blockIdx.x & 31) << 6;           // 64 columns per block
    int t  = threadIdx.x;
    int c  = m0 + (t & 63);
    int r0 = t >> 6;
    const float* Sb = S + (size_t)b * Nn * Nn;
    float cm = fdec(keys[(size_t)b * Nn + c]);
    float s = 0.f;
    for (int n = r0; n < Nn; n += 4)
        s += expf(Sb[(size_t)n * Nn + c] - cm);
    __shared__ float red[256];
    red[t] = s;
    __syncthreads();
    if (t < 64) {
        float tot = red[t] + red[t + 64] + red[t + 128] + red[t + 192];
        colrs[(size_t)b * Nn + m0 + t] = 1.0f / tot;
        ((float*)keys)[(size_t)b * Nn + m0 + t] = cm;   // overwrite key with decoded float max
    }
}

// ---------- P = exp(S - colmax)/colsum, split to bf16 hi/lo ----------
__global__ void p_split_kernel(const float* __restrict__ S, const float* __restrict__ cmf,
                               const float* __restrict__ colrs,
                               u16* __restrict__ ph, u16* __restrict__ pl) {
    long long i = (long long)blockIdx.x * blockDim.x + threadIdx.x;
    long long stride = (long long)gridDim.x * blockDim.x;
    for (; i < BNN; i += stride) {
        int b = (int)(i >> 22);                 // Nn*Nn = 2^22
        int m = (int)(i & 2047);
        size_t cidx = (size_t)b * Nn + m;
        float p = expf(S[i] - cmf[cidx]) * colrs[cidx];
        u16 h = f2bf(p);
        ph[i] = h; pl[i] = f2bf(p - bf2f(h));
    }
}

extern "C" void kernel_launch(void* const* d_in, const int* in_sizes, int n_in,
                              void* d_out, int out_size, void* d_ws, size_t ws_size,
                              hipStream_t stream)
{
    const float* img  = (const float*)d_in[0];
    const float* text = (const float*)d_in[1];
    const float* Wq   = (const float*)d_in[2];
    const float* Wk   = (const float*)d_in[3];
    const float* Wv   = (const float*)d_in[4];
    float* out  = (float*)d_out;
    float* feat = out + BND;

    char* ws = (char*)d_ws;
    const size_t BND_BF = (size_t)BND * 2;   // 33,554,432 B per split array
    const size_t W_BF   = (size_t)Dd * Dd * 2;

    // [0, 134MB): img/text splits, later reused as S (fp32, B*N*N)
    u16*   img_hi  = (u16*)(ws + 0 * BND_BF);
    u16*   img_lo  = (u16*)(ws + 1 * BND_BF);
    u16*   text_hi = (u16*)(ws + 2 * BND_BF);
    u16*   text_lo = (u16*)(ws + 3 * BND_BF);
    float* S       = (float*)(ws + 0);
    size_t o = 4 * BND_BF;
    u16* wq_hi = (u16*)(ws + o); o += W_BF;
    u16* wq_lo = (u16*)(ws + o); o += W_BF;
    u16* wk_hi = (u16*)(ws + o); o += W_BF;
    u16* wk_lo = (u16*)(ws + o); o += W_BF;
    u16* wv_hi = (u16*)(ws + o); o += W_BF;
    u16* wv_lo = (u16*)(ws + o); o += W_BF;
    size_t q_off = o;
    u16* q_hi  = (u16*)(ws + o); o += BND_BF;
    u16* q_lo  = (u16*)(ws + o); o += BND_BF;
    size_t k_off = o;
    u16* k_hi  = (u16*)(ws + o); o += BND_BF;
    u16* k_lo  = (u16*)(ws + o); o += BND_BF;
    u16* vt_hi = (u16*)(ws + o); o += BND_BF;
    u16* vt_lo = (u16*)(ws + o); o += BND_BF;
    u32*   cmax  = (u32*)(ws + o);   o += (size_t)Bb * Nn * 4;
    float* colrs = (float*)(ws + o); o += (size_t)Bb * Nn * 4;
    // P splits reuse the (now dead) Q/K split regions: 2 x 67MB
    u16* p_hi = (u16*)(ws + q_off);
    u16* p_lo = (u16*)(ws + k_off);

    // 1) split fp32 inputs into bf16 hi/lo
    split_kernel<<<2048, 256, 0, stream>>>(img,  img_hi,  img_lo,  BND);
    split_kernel<<<2048, 256, 0, stream>>>(text, text_hi, text_lo, BND);
    split_kernel<<<512,  256, 0, stream>>>(Wq, wq_hi, wq_lo, (long long)Dd * Dd);
    split_kernel<<<512,  256, 0, stream>>>(Wk, wk_hi, wk_lo, (long long)Dd * Dd);
    split_kernel<<<512,  256, 0, stream>>>(Wv, wv_hi, wv_lo, (long long)Dd * Dd);

    // 2) projections: Q = img@Wq^T, K = text@Wk^T (row-major), V stored transposed
    gemm_bf16x3<MODE_SPLIT><<<dim3(128, 16, 1), 256, 0, stream>>>(
        img_hi, img_lo, Dd, 0, wq_hi, wq_lo, Dd, 0, Dd, Dd, 0,
        nullptr, q_hi, q_lo, nullptr, nullptr, nullptr);
    gemm_bf16x3<MODE_SPLIT><<<dim3(128, 16, 1), 256, 0, stream>>>(
        text_hi, text_lo, Dd, 0, wk_hi, wk_lo, Dd, 0, Dd, Dd, 0,
        nullptr, k_hi, k_lo, nullptr, nullptr, nullptr);
    gemm_bf16x3<MODE_SPLIT_T><<<dim3(128, 16, 1), 256, 0, stream>>>(
        text_hi, text_lo, Dd, 0, wv_hi, wv_lo, Dd, 0, Dd, Dd, 0,
        nullptr, vt_hi, vt_lo, nullptr, nullptr, nullptr);

    // 3) S = Q@K^T per batch, with per-column (over n) max via keyed atomicMax
    fill_u32<<<64, 256, 0, stream>>>(cmax, 0u, Bb * Nn);
    gemm_bf16x3<MODE_SCORE><<<dim3(16, 32, Bb), 256, 0, stream>>>(
        q_hi, q_lo, Dd, (long long)Nn * Dd, k_hi, k_lo, Dd, (long long)Nn * Dd,
        Dd, Nn, (long long)Nn * Nn,
        S, nullptr, nullptr, cmax, nullptr, nullptr);

    // 4) column softmax stats: colrs = 1/sum_n exp(S - max), decode max to float
    colsum_kernel<<<Bb * 32, 256, 0, stream>>>(S, cmax, colrs);

    // 5) P = exp(S - max)*colrs, split to bf16 hi/lo (overwrites Q/K splits)
    p_split_kernel<<<8192, 256, 0, stream>>>(S, (const float*)cmax, colrs, p_hi, p_lo);

    // 6) out = P @ V (B-fragments contiguous from Vt), feature = out + text
    gemm_bf16x3<MODE_FINAL><<<dim3(16, 16, Bb), 256, 0, stream>>>(
        p_hi, p_lo, Nn, (long long)Nn * Nn, vt_hi, vt_lo, Nn, (long long)Dd * Nn,
        Nn, Dd, 0,
        out, nullptr, nullptr, nullptr, text, feat);

    (void)in_sizes; (void)n_in; (void)out_size; (void)ws_size;
}
